// ComplexAttention_38027640438965
// MI455X (gfx1250) — matrix operand order repaired and checked by reading, not compiled, not run on hardware
//
#include <hip/hip_runtime.h>

// Problem constants (from reference)
#define BB 2
#define LL 2048
#define DD 1024
#define HH 16
#define HD 64
// SCALE = HD^-0.5 = 0.125

typedef __attribute__((ext_vector_type(16))) _Float16 v16h;
typedef __attribute__((ext_vector_type(8)))  _Float16 v8h;
typedef __attribute__((ext_vector_type(4)))  _Float16 v4h;
typedef __attribute__((ext_vector_type(8)))  float    v8f;
typedef __attribute__((ext_vector_type(4)))  float    v4f;
typedef __attribute__((ext_vector_type(4)))  unsigned int v4u;
typedef __attribute__((ext_vector_type(8)))  int      v8i;
typedef __attribute__((ext_vector_type(4)))  int      v4i;

// D = A(16x32 f16) * B(32x16 f16) + C(16x16 f32), f32 accumulate
__device__ __forceinline__ v8f wmma16(v16h a, v16h b, v8f c) {
  return __builtin_amdgcn_wmma_f32_16x16x32_f16(
      /*neg_a=*/false, a, /*neg_b=*/false, b,
      /*c_mod=*/(short)0, c, /*reuse_a=*/false, /*reuse_b=*/false);
}

union HV { v16h v; v8h h[2]; };

// A fragment 16x32 (MxK), row-major source with K contiguous.
__device__ __forceinline__ v16h ldA(const _Float16* p, int stride) {
  const int lane = threadIdx.x & 31;
  const int m  = lane & 15;
  const int kb = (lane >> 4) << 3;
  HV u;
  u.h[0] = *(const v8h*)(p + m * stride + kb);
  u.h[1] = *(const v8h*)(p + m * stride + kb + 16);
  return u.v;
}

// B fragment 32x16 (KxN) for A@B^T: B[k][n] = p[n*stride + k] (K contiguous).
__device__ __forceinline__ v16h ldB(const _Float16* p, int stride) {
  const int lane = threadIdx.x & 31;
  const int n  = lane & 15;
  const int kb = (lane >> 4) << 3;
  HV u;
  u.h[0] = *(const v8h*)(p + n * stride + kb);
  u.h[1] = *(const v8h*)(p + n * stride + kb + 16);
  return u.v;
}

// ---------------------------------------------------------------------------
// TDM: load a 64x64 f16 tile (row stride `row_stride` elems) into LDS with
// +8-half row padding (pad_interval = 32 DWORDs, pad_amount = 4 DWORDs),
// producing rows of KSTR=72 halves. Descriptor per CDNA5 D# (08_async_tensor).
// ---------------------------------------------------------------------------
#define KSTR 72   // 64 + 8 pad halves

__device__ __forceinline__ void tdm_load_tile(unsigned lds_off, const void* gsrc,
                                              unsigned row_stride)
{
#if __has_builtin(__builtin_amdgcn_tensor_load_to_lds)
  unsigned long long ga = (unsigned long long)(uintptr_t)gsrc;
  // Group 0: count=1 (valid user descriptor), lds_addr, global_addr[56:0], type=2
  v4u g0 = { 1u,
             lds_off,
             (unsigned)ga,
             (unsigned)((ga >> 32) & 0x01ffffffu) | (2u << 30) };
  // Group 1
  v8i g1;
  g1[0] = (int)((1u << 16)     // data_size = 2 bytes
              | (1u << 20)     // pad_enable
              | (4u << 22)     // pad_interval: 32 DWORDs (one 64-half row)
              | (3u << 25));   // pad_amount: 4 DWORDs (8 halves)
  g1[1] = (int)((row_stride & 0xffffu) << 16);          // tensor_dim0[15:0]
  g1[2] = (int)((row_stride >> 16) & 0xffffu);          // tensor_dim0[31:16]; tensor_dim1[15:0]=0
  g1[3] = (int)(1u | (64u << 16));                      // tensor_dim1[31:16]=1 (dim1=65536); tile_dim0=64
  g1[4] = 64;                                           // tile_dim1=64, tile_dim2=0
  g1[5] = (int)row_stride;                              // tensor_dim0_stride[31:0]
  g1[6] = 0;                                            // stride hi / dim1_stride lo
  g1[7] = 0;
  v4i g2 = {}, g3 = {};
#if defined(__clang_major__) && __clang_major__ >= 23
  v8i g4 = {};
  __builtin_amdgcn_tensor_load_to_lds(g0, g1, g2, g3, g4, 0);
#else
  __builtin_amdgcn_tensor_load_to_lds(g0, g1, g2, g3, 0);
#endif
#else
  (void)lds_off; (void)gsrc; (void)row_stride;
#endif
}

__device__ __forceinline__ void tdm_wait() {
#if __has_builtin(__builtin_amdgcn_s_wait_tensorcnt)
  __builtin_amdgcn_s_wait_tensorcnt((short)0);
#else
  asm volatile("s_wait_tensorcnt 0x0" ::: "memory");
#endif
}

// ---------------------------------------------------------------------------
// Kernel 1: complex projection  out = (xr + i*xi) @ (wr + i*wi)^T  -> f16
//   mode 0: store [B,H,L,HD] (Q, K) ; mode 1: store transposed [B,H,HD,L] (V)
// ---------------------------------------------------------------------------
#define PK   32
#define PSTR 40
#define PM   64
#define PN   128

__global__ __launch_bounds__(256)
void cproj_kernel(const float* __restrict__ xr, const float* __restrict__ xi,
                  const float* __restrict__ wr, const float* __restrict__ wi,
                  _Float16* __restrict__ outR, _Float16* __restrict__ outI, int mode)
{
  __shared__ __align__(16) _Float16 sxr[2][PM * PSTR], sxi[2][PM * PSTR];
  __shared__ __align__(16) _Float16 swr[2][PN * PSTR], swi[2][PN * PSTR];

  const int t    = threadIdx.x;
  const int m0   = blockIdx.y * PM;
  const int n0   = blockIdx.x * PN;
  const int wave = t >> 5;
  const int wm   = (wave & 3) * 16;
  const int wn   = (wave >> 2) * 64;

  v8f accR[4] = {}, accI[4] = {};

  auto stage = [&](int kk, int buf) {
    const int k0 = kk * PK;
#pragma unroll
    for (int i = 0; i < 2; ++i) {          // x tiles: 64 rows x 32 (r and i)
      int idx = t + i * 256;
      int row = idx >> 3, c4 = (idx & 7) * 4;
      v4f fr = *(const v4f*)(xr + (long)(m0 + row) * DD + k0 + c4);
      v4f fi = *(const v4f*)(xi + (long)(m0 + row) * DD + k0 + c4);
      v4h hr, hi;
#pragma unroll
      for (int e = 0; e < 4; ++e) { hr[e] = (_Float16)fr[e]; hi[e] = (_Float16)fi[e]; }
      *(v4h*)(&sxr[buf][row * PSTR + c4]) = hr;
      *(v4h*)(&sxi[buf][row * PSTR + c4]) = hi;
    }
#pragma unroll
    for (int i = 0; i < 4; ++i) {          // w tiles: 128 rows x 32 (r and i)
      int idx = t + i * 256;
      int row = idx >> 3, c4 = (idx & 7) * 4;
      v4f fr = *(const v4f*)(wr + (long)(n0 + row) * DD + k0 + c4);
      v4f fi = *(const v4f*)(wi + (long)(n0 + row) * DD + k0 + c4);
      v4h hr, hi;
#pragma unroll
      for (int e = 0; e < 4; ++e) { hr[e] = (_Float16)fr[e]; hi[e] = (_Float16)fi[e]; }
      *(v4h*)(&swr[buf][row * PSTR + c4]) = hr;
      *(v4h*)(&swi[buf][row * PSTR + c4]) = hi;
    }
  };

  stage(0, 0);
  __syncthreads();
  const int KT = DD / PK;
  for (int kk = 0; kk < KT; ++kk) {
    const int buf = kk & 1;
    if (kk + 1 < KT) stage(kk + 1, buf ^ 1);

    v16h ar = ldA(&sxr[buf][wm * PSTR], PSTR);
    v16h ai = ldA(&sxi[buf][wm * PSTR], PSTR);
    v16h an = -ai;                          // real part: Ar@Wr^T + (-Ai)@Wi^T
#pragma unroll
    for (int tt = 0; tt < 4; ++tt) {
      v16h br = ldB(&swr[buf][(wn + tt * 16) * PSTR], PSTR);
      v16h bi = ldB(&swi[buf][(wn + tt * 16) * PSTR], PSTR);
      accR[tt] = wmma16(ar, br, accR[tt]);
      accR[tt] = wmma16(an, bi, accR[tt]);
      accI[tt] = wmma16(ar, bi, accI[tt]);
      accI[tt] = wmma16(ai, br, accI[tt]);
    }
    __syncthreads();
  }

  const int lane = t & 31;
  const int mrow = m0 + wm + ((lane >> 4) << 3);
#pragma unroll
  for (int tt = 0; tt < 4; ++tt) {
    const int d  = n0 + wn + tt * 16 + (lane & 15);
    const int h  = d >> 6, hd = d & 63;
#pragma unroll
    for (int j = 0; j < 8; ++j) {
      const int m = mrow + j;
      const int b = m >> 11, lq = m & 2047;
      long addr = (mode == 0)
        ? (((long)(b * HH + h)) * LL + lq) * HD + hd
        : (((long)(b * HH + h)) * HD + hd) * (long)LL + lq;
      outR[addr] = (_Float16)accR[tt][j];
      outI[addr] = (_Float16)accI[tt][j];
    }
  }
}

// ---------------------------------------------------------------------------
// Kernel 2: causal flash attention, TDM-staged K/V tiles, double-buffered.
// Block: 128 q-rows (8 waves x 16), key blocks of 64.
// ---------------------------------------------------------------------------
__global__ __launch_bounds__(256)
void cattn_kernel(const _Float16* __restrict__ Qr, const _Float16* __restrict__ Qi,
                  const _Float16* __restrict__ Kr, const _Float16* __restrict__ Ki,
                  const _Float16* __restrict__ Vtr, const _Float16* __restrict__ Vti,
                  _Float16* __restrict__ Or, _Float16* __restrict__ Oi)
{
  // [buf][Kr,Ki,Vr,Vi][64 rows x KSTR]  (2 x 4 x 9216B = 72KB) + P (18KB)
  __shared__ __align__(16) _Float16 skv[2][4][64 * KSTR];
  __shared__ __align__(16) _Float16 sp[8][16 * KSTR];

  const int t = threadIdx.x, lane = t & 31, wave = t >> 5;
  const int b = blockIdx.z, h = blockIdx.y, qt = blockIdx.x;
  const int q0 = qt * 128 + wave * 16;
  const long bh = (long)(b * HH + h);

  const _Float16* qrp = Qr + (bh * LL + q0) * HD;
  const _Float16* qip = Qi + (bh * LL + q0) * HD;
  const _Float16* krp = Kr + bh * LL * HD;
  const _Float16* kip = Ki + bh * LL * HD;
  const _Float16* vrp = Vtr + bh * HD * (long)LL;   // transposed [HD, L]
  const _Float16* vip = Vti + bh * HD * (long)LL;

  // TDM issue: waves 0..3 each own one of the four 64x64 tiles.
  auto issue = [&](int kb, int buf) {
    const int key0 = kb * 64;
    const _Float16* src;
    unsigned stride;
    if      (wave == 0) { src = krp + (long)key0 * HD; stride = HD; }
    else if (wave == 1) { src = kip + (long)key0 * HD; stride = HD; }
    else if (wave == 2) { src = vrp + key0;            stride = LL; }
    else                { src = vip + key0;            stride = LL; }
    unsigned lds_off = (unsigned)(uintptr_t)(const void*)&skv[buf][wave][0];
    tdm_load_tile(lds_off, src, stride);
  };

  // Q fragments (resident), pre-scaled by HD^-0.5
  const _Float16 sc = (_Float16)0.125f;
  v16h qr0 = ldA(qrp, HD) * sc, qr1 = ldA(qrp + 32, HD) * sc;
  v16h qi0 = ldA(qip, HD) * sc, qi1 = ldA(qip + 32, HD) * sc;

  v8f oR[4] = {}, oI[4] = {};
  float mrow[8], lrow[8];
#pragma unroll
  for (int j = 0; j < 8; ++j) { mrow[j] = -3.0e38f; lrow[j] = 0.f; }

  const int nkb = qt * 2 + 2;   // key blocks covering 0 .. qt*128+127
  if (wave < 4) { issue(0, 0); tdm_wait(); }
  __syncthreads();

  for (int kb = 0; kb < nkb; ++kb) {
    const int key0 = kb * 64;
    const int buf  = kb & 1;
    if (wave < 4 && kb + 1 < nkb) issue(kb + 1, buf ^ 1);   // prefetch via TDM

    if (key0 <= q0 + 15) {   // this wave has live keys in this block
      const _Float16* skr = &skv[buf][0][0];
      const _Float16* ski = &skv[buf][1][0];
      const _Float16* svr = &skv[buf][2][0];
      const _Float16* svi = &skv[buf][3][0];

      // S = Qr@Kr^T + Qi@Ki^T  (16 WMMAs)
      v8f s[4] = {};
#pragma unroll
      for (int tt = 0; tt < 4; ++tt) {
        const _Float16* kr = &skr[(tt * 16) * KSTR];
        const _Float16* ki = &ski[(tt * 16) * KSTR];
        s[tt] = wmma16(qr0, ldB(kr, KSTR), s[tt]);
        s[tt] = wmma16(qr1, ldB(kr + 32, KSTR), s[tt]);
        s[tt] = wmma16(qi0, ldB(ki, KSTR), s[tt]);
        s[tt] = wmma16(qi1, ldB(ki + 32, KSTR), s[tt]);
      }

      // Causal mask + online softmax. Lane owns rows M=8*(lane>=16)+j, col N=lane&15.
      const int keybase = key0 + (lane & 15);
      const int qrow    = q0 + ((lane >> 4) << 3);
      float mnew[8], al[8];
#pragma unroll
      for (int j = 0; j < 8; ++j) {
        float rmax = -3.0e38f;
#pragma unroll
        for (int tt = 0; tt < 4; ++tt) {
          float v = (keybase + tt * 16 <= qrow + j) ? s[tt][j] : -3.0e38f;
          s[tt][j] = v;
          rmax = fmaxf(rmax, v);
        }
        rmax = fmaxf(rmax, __shfl_xor(rmax, 1));
        rmax = fmaxf(rmax, __shfl_xor(rmax, 2));
        rmax = fmaxf(rmax, __shfl_xor(rmax, 4));
        rmax = fmaxf(rmax, __shfl_xor(rmax, 8));
        mnew[j] = fmaxf(mrow[j], rmax);
        al[j]   = __expf(mrow[j] - mnew[j]);
        mrow[j] = mnew[j];
      }
#pragma unroll
      for (int j = 0; j < 8; ++j) {
        float rsum = 0.f;
#pragma unroll
        for (int tt = 0; tt < 4; ++tt) {
          float p = __expf(s[tt][j] - mnew[j]);
          s[tt][j] = p;
          rsum += p;
        }
        rsum += __shfl_xor(rsum, 1);
        rsum += __shfl_xor(rsum, 2);
        rsum += __shfl_xor(rsum, 4);
        rsum += __shfl_xor(rsum, 8);
        lrow[j] = lrow[j] * al[j] + rsum;
#pragma unroll
        for (int tt = 0; tt < 4; ++tt) { oR[tt][j] *= al[j]; oI[tt][j] *= al[j]; }
      }

      // C-layout -> A-layout through wave-private LDS P buffer (f16)
      const int prow = (lane >> 4) << 3;
#pragma unroll
      for (int tt = 0; tt < 4; ++tt)
#pragma unroll
        for (int j = 0; j < 8; ++j)
          sp[wave][(prow + j) * KSTR + tt * 16 + (lane & 15)] = (_Float16)s[tt][j];
      asm volatile("s_wait_dscnt 0x0" ::: "memory");  // wave-local LDS RAW

      // O += P @ V  (16 WMMAs)
#pragma unroll
      for (int ks = 0; ks < 2; ++ks) {
        v16h pa = ldA(&sp[wave][ks * 32], KSTR);
#pragma unroll
        for (int tt = 0; tt < 4; ++tt) {
          oR[tt] = wmma16(pa, ldB(&svr[(tt * 16) * KSTR + ks * 32], KSTR), oR[tt]);
          oI[tt] = wmma16(pa, ldB(&svi[(tt * 16) * KSTR + ks * 32], KSTR), oI[tt]);
        }
      }
    }

    if (wave < 4 && kb + 1 < nkb) tdm_wait();   // next tiles landed
    __syncthreads();
  }

  // Normalize and store f16 O in [B, L, D] (D = h*64 + hd).
  float rl[8];
#pragma unroll
  for (int j = 0; j < 8; ++j) rl[j] = 1.0f / lrow[j];
  const int qbase = q0 + ((lane >> 4) << 3);
#pragma unroll
  for (int tt = 0; tt < 4; ++tt) {
    const int d = h * HD + tt * 16 + (lane & 15);
#pragma unroll
    for (int j = 0; j < 8; ++j) {
      long addr = ((long)b * LL + qbase + j) * DD + d;
      Or[addr] = (_Float16)(oR[tt][j] * rl[j]);
      Oi[addr] = (_Float16)(oI[tt][j] * rl[j]);
    }
  }
}

// ---------------------------------------------------------------------------
// Kernel 3: output complex GEMM + bias, f32 store to d_out.
// ---------------------------------------------------------------------------
__global__ __launch_bounds__(256)
void cout_kernel(const _Float16* __restrict__ Ar, const _Float16* __restrict__ Ai,
                 const float* __restrict__ wr, const float* __restrict__ wi,
                 const float* __restrict__ br_, const float* __restrict__ bi_,
                 float* __restrict__ yr, float* __restrict__ yi)
{
  __shared__ __align__(16) _Float16 sar[2][PM * PSTR], sai[2][PM * PSTR];
  __shared__ __align__(16) _Float16 swr[2][PN * PSTR], swi[2][PN * PSTR];

  const int t    = threadIdx.x;
  const int m0   = blockIdx.y * PM;
  const int n0   = blockIdx.x * PN;
  const int wave = t >> 5;
  const int wm   = (wave & 3) * 16;
  const int wn   = (wave >> 2) * 64;

  v8f accR[4] = {}, accI[4] = {};

  auto stage = [&](int kk, int buf) {
    const int k0 = kk * PK;
    {                                   // A tiles already f16: straight copy
      int row = t >> 2, c = (t & 3) * 8;
      *(v8h*)(&sar[buf][row * PSTR + c]) = *(const v8h*)(Ar + (long)(m0 + row) * DD + k0 + c);
      *(v8h*)(&sai[buf][row * PSTR + c]) = *(const v8h*)(Ai + (long)(m0 + row) * DD + k0 + c);
    }
#pragma unroll
    for (int i = 0; i < 4; ++i) {       // W tiles fp32 -> f16
      int idx = t + i * 256;
      int row = idx >> 3, c4 = (idx & 7) * 4;
      v4f fr = *(const v4f*)(wr + (long)(n0 + row) * DD + k0 + c4);
      v4f fi = *(const v4f*)(wi + (long)(n0 + row) * DD + k0 + c4);
      v4h hr, hi;
#pragma unroll
      for (int e = 0; e < 4; ++e) { hr[e] = (_Float16)fr[e]; hi[e] = (_Float16)fi[e]; }
      *(v4h*)(&swr[buf][row * PSTR + c4]) = hr;
      *(v4h*)(&swi[buf][row * PSTR + c4]) = hi;
    }
  };

  stage(0, 0);
  __syncthreads();
  const int KT = DD / PK;
  for (int kk = 0; kk < KT; ++kk) {
    const int buf = kk & 1;
    if (kk + 1 < KT) stage(kk + 1, buf ^ 1);
    v16h ar = ldA(&sar[buf][wm * PSTR], PSTR);
    v16h ai = ldA(&sai[buf][wm * PSTR], PSTR);
    v16h an = -ai;
#pragma unroll
    for (int tt = 0; tt < 4; ++tt) {
      v16h br = ldB(&swr[buf][(wn + tt * 16) * PSTR], PSTR);
      v16h bi = ldB(&swi[buf][(wn + tt * 16) * PSTR], PSTR);
      accR[tt] = wmma16(ar, br, accR[tt]);
      accR[tt] = wmma16(an, bi, accR[tt]);
      accI[tt] = wmma16(ar, bi, accI[tt]);
      accI[tt] = wmma16(ai, br, accI[tt]);
    }
    __syncthreads();
  }

  const int lane = t & 31;
  const int mrow = m0 + wm + ((lane >> 4) << 3);
#pragma unroll
  for (int tt = 0; tt < 4; ++tt) {
    const int d  = n0 + wn + tt * 16 + (lane & 15);
    const float vb_r = br_[d], vb_i = bi_[d];
#pragma unroll
    for (int j = 0; j < 8; ++j) {
      const long m = mrow + j;
      yr[m * DD + d] = accR[tt][j] + vb_r;
      yi[m * DD + d] = accI[tt][j] + vb_i;
    }
  }
}

// ---------------------------------------------------------------------------
extern "C" void kernel_launch(void* const* d_in, const int* in_sizes, int n_in,
                              void* d_out, int out_size, void* d_ws, size_t ws_size,
                              hipStream_t stream)
{
  (void)in_sizes; (void)n_in; (void)out_size; (void)ws_size;
  const float* xr   = (const float*)d_in[0];
  const float* xi   = (const float*)d_in[1];
  const float* wq_r = (const float*)d_in[2];
  const float* wq_i = (const float*)d_in[3];
  const float* wk_r = (const float*)d_in[4];
  const float* wk_i = (const float*)d_in[5];
  const float* wv_r = (const float*)d_in[6];
  const float* wv_i = (const float*)d_in[7];
  const float* wo_r = (const float*)d_in[8];
  const float* wo_i = (const float*)d_in[9];
  const float* bo_r = (const float*)d_in[10];
  const float* bo_i = (const float*)d_in[11];

  const long S = (long)BB * LL * DD;          // 4,194,304 elements
  _Float16* ws = (_Float16*)d_ws;             // 8*S halves = 64 MiB scratch
  _Float16 *Qr = ws,         *Qi  = ws + S,
           *Kr = ws + 2 * S, *Ki  = ws + 3 * S,
           *Vtr = ws + 4 * S, *Vti = ws + 5 * S,
           *Or = ws + 6 * S, *Oi  = ws + 7 * S;
  float* yr = (float*)d_out;
  float* yi = yr + S;

  dim3 gg(DD / PN, (BB * LL) / PM);           // 8 x 64 blocks
  cproj_kernel<<<gg, 256, 0, stream>>>(xr, xi, wq_r, wq_i, Qr, Qi, 0);
  cproj_kernel<<<gg, 256, 0, stream>>>(xr, xi, wk_r, wk_i, Kr, Ki, 0);
  cproj_kernel<<<gg, 256, 0, stream>>>(xr, xi, wv_r, wv_i, Vtr, Vti, 1);

  dim3 ga(LL / 128, HH, BB);                  // 16 x 16 x 2 blocks
  cattn_kernel<<<ga, 256, 0, stream>>>(Qr, Qi, Kr, Ki, Vtr, Vti, Or, Oi);

  cout_kernel<<<gg, 256, 0, stream>>>(Or, Oi, wo_r, wo_i, bo_r, bo_i, yr, yi);
}
